// DualPassDeltaNetLayer_70205535420736
// MI455X (gfx1250) — matrix operand order repaired and checked by reading, not compiled, not run on hardware
//
#include <hip/hip_runtime.h>
#include <hip/hip_bf16.h>

// ---- problem constants (fixed by the reference) ----
#define B_     2
#define L_     4096
#define D_     1024
#define H_     16
#define HD_    64
#define CHUNK_ 128
#define M_     (B_ * L_)        // 8192 rows
#define NCHUNK (L_ / CHUNK_)    // 32
#define BH     (B_ * H_)        // 32

typedef __bf16 bf16;
typedef __attribute__((ext_vector_type(16))) __bf16 v16bf;
typedef __attribute__((ext_vector_type(8)))  __bf16 bf8v;
typedef __attribute__((ext_vector_type(4)))  __bf16 bf4;
typedef __attribute__((ext_vector_type(8)))  float  v8f;
typedef __attribute__((ext_vector_type(4)))  unsigned int v4u;
typedef __attribute__((ext_vector_type(8)))  int v8i;
typedef __attribute__((ext_vector_type(4)))  int v4i;

// ===== staging-path selection (probed: clang-23, tensor_load_to_lds 6-arg) ===
#if defined(__gfx1250__)
#if __has_builtin(__builtin_amdgcn_tensor_load_to_lds_d2)
#define TDM_MODE 2
#elif __has_builtin(__builtin_amdgcn_tensor_load_to_lds)
#define TDM_MODE 1
#else
#define TDM_MODE 0
#endif
#if __has_builtin(__builtin_amdgcn_global_load_async_to_lds_b128)
#define ASYNC_MODE 1
#else
#define ASYNC_MODE 0
#endif
#else
#define TDM_MODE 0
#define ASYNC_MODE 0
#endif

#ifdef __AMDGCN__
typedef __attribute__((address_space(1))) v4i* as1_v4i;  // AS1 prints as __device__
typedef __attribute__((address_space(3))) v4i* as3_v4i;  // AS3 prints as __shared__
#endif

__device__ __forceinline__ void wait_tensor0() {
#if TDM_MODE
#if __has_builtin(__builtin_amdgcn_s_wait_tensorcnt)
  __builtin_amdgcn_s_wait_tensorcnt(0);
#else
  asm volatile("s_wait_tensorcnt 0x0" ::: "memory");
#endif
#endif
}

__device__ __forceinline__ void wait_async0() {
#if ASYNC_MODE && !TDM_MODE
#if __has_builtin(__builtin_amdgcn_s_wait_asynccnt)
  __builtin_amdgcn_s_wait_asynccnt(0);
#else
  asm volatile("s_wait_asynccnt 0x0" ::: "memory");
#endif
#endif
}

__device__ __forceinline__ float sigm(float x) { return 1.0f / (1.0f + __expf(-x)); }

__device__ __forceinline__ float wred(float v) {
#pragma unroll
  for (int m = 16; m > 0; m >>= 1) v += __shfl_xor(v, m, 32);
  return v;
}

// Build a 16-element bf16 fragment from two contiguous 16-byte LDS runs.
__device__ __forceinline__ v16bf ld_frag(const bf16* p0, const bf16* p1) {
  bf8v lo = *(const bf8v*)p0;
  bf8v hi = *(const bf8v*)p1;
  return __builtin_shufflevector(lo, hi, 0, 1, 2, 3, 4, 5, 6, 7, 8, 9, 10, 11, 12, 13, 14, 15);
}

// ---------------------------------------------------------------------------
// Tensor Data Mover: async 2D tile (16-bit elems) global -> LDS with row pad.
//   td0 = tile width (elems), td1 = tile rows, stride0 = global row stride,
//   pint/pamt = D# pad_interval / pad_amount encodings (LDS row padding).
// Issued once per wave (EXEC ignored by TDM); tracked by TENSORcnt.
// ---------------------------------------------------------------------------
__device__ __forceinline__ void tdm_load_2d(const void* gaddr, const void* ldsaddr,
                                            unsigned td0, unsigned td1, unsigned stride0,
                                            unsigned pint, unsigned pamt) {
#if TDM_MODE
  const unsigned long long ga = (unsigned long long)gaddr;
  const unsigned la = (unsigned)(unsigned long long)ldsaddr;  // LDS byte offset
  v4u g0;
  g0[0] = 1u;                                            // count=1, user descriptor
  g0[1] = la;                                            // lds_addr
  g0[2] = (unsigned)ga;                                  // global_addr[31:0]
  g0[3] = (unsigned)((ga >> 32) & 0x01FFFFFFull) | 0x80000000u;  // addr[56:32] | type=2
  v8i g1;
  g1[0] = (int)((1u << 16) |                             // data_size = 2 bytes
                (1u << 20) |                             // pad_enable
                ((pint & 7u) << 22) | ((pamt & 127u) << 25));
  g1[1] = (int)((td0 & 0xFFFFu) << 16);                  // tensor_dim0[15:0]
  g1[2] = (int)(((td0 >> 16) & 0xFFFFu) | ((td1 & 0xFFFFu) << 16));  // dim0 hi | dim1 lo
  g1[3] = (int)(((td1 >> 16) & 0xFFFFu) | ((td0 & 0xFFFFu) << 16));  // dim1 hi | tile_dim0
  g1[4] = (int)(td1 & 0xFFFFu);                          // tile_dim1 (tile_dim2 = 0)
  g1[5] = (int)stride0;                                  // tensor_dim0_stride
  g1[6] = 0;
  g1[7] = 0;
#if TDM_MODE == 2
  __builtin_amdgcn_tensor_load_to_lds_d2(g0, g1, 0);
#else
  const v4i z = {0, 0, 0, 0};
#if __clang_major__ >= 23
  const v8i z8 = {0, 0, 0, 0, 0, 0, 0, 0};
  __builtin_amdgcn_tensor_load_to_lds(g0, g1, z, z, z8, 0);
#else
  __builtin_amdgcn_tensor_load_to_lds(g0, g1, z, z, 0);
#endif
#endif
#else
  (void)gaddr; (void)ldsaddr; (void)td0; (void)td1; (void)stride0; (void)pint; (void)pamt;
#endif
}

// ---------------------------------------------------------------------------
// Async global->LDS tile staging (per-lane GLOBAL_LOAD_ASYNC_TO_LDS_B128).
// Tier-2 path, only compiled if the TDM tier is unavailable.
// ---------------------------------------------------------------------------
template <int ROWS, int CHUNKS, int STRIDE, int NT>
__device__ __forceinline__ void async_stage_tile(const bf16* g, int gstride,
                                                 bf16* ldsbase, int tid) {
#if ASYNC_MODE && !TDM_MODE
  for (int i = tid; i < ROWS * CHUNKS; i += NT) {
    const int r = i / CHUNKS, c = (i % CHUNKS) * 8;
    const bf16* gp = g + (size_t)r * gstride + c;
    bf16* lp = ldsbase + r * STRIDE + c;
    __builtin_amdgcn_global_load_async_to_lds_b128(
        (as1_v4i)(unsigned long long)gp, (as3_v4i)(unsigned long long)lp, 0, 0);
  }
#else
  (void)g; (void)gstride; (void)ldsbase; (void)tid;
#endif
}

// ---------------------------------------------------------------------------
// fp32 -> bf16 conversion (grid stride)
// ---------------------------------------------------------------------------
__global__ void cvt_f32_to_bf16(const float* __restrict__ s, bf16* __restrict__ d, int n) {
  for (int i = blockIdx.x * blockDim.x + threadIdx.x; i < n; i += gridDim.x * blockDim.x)
    d[i] = (bf16)s[i];
}

// ---------------------------------------------------------------------------
// fp32 [K][N] -> bf16 transposed [N][K]  (LDS-tiled 32x32)
// ---------------------------------------------------------------------------
__global__ __launch_bounds__(256) void cvt_transpose_bf16(const float* __restrict__ s,
                                                          bf16* __restrict__ d) {
  __shared__ float tile[32][33];
  const int bx = blockIdx.x * 32, by = blockIdx.y * 32;
  const int tx = threadIdx.x & 31, ty = threadIdx.x >> 5;
  for (int i = ty; i < 32; i += 8) tile[i][tx] = s[(size_t)(by + i) * D_ + bx + tx];
  __syncthreads();
  for (int i = ty; i < 32; i += 8) d[(size_t)(bx + i) * D_ + by + tx] = (bf16)tile[tx][i];
}

// ---------------------------------------------------------------------------
// WMMA GEMM: C[M_,D_] = A[M_,D_] @ W  with W supplied TRANSPOSED (Wt[n][k]).
//   MODE 0: epilogue = SiLU, store bf16 in [B,H,L,HD] head layout
//   MODE 1: plain fp32 store [M_,D_]
// Block tile 128x128, 8 waves (4x2), wave tile 32x64 (2x4 wmma), K-step 32.
// Tiles double-buffered via TDM (preferred) / async-LDS / plain copies.
// ---------------------------------------------------------------------------
template <int MODE>
__global__ __launch_bounds__(256) void gemm_wmma(
    const bf16* __restrict__ A, const bf16* __restrict__ Wt,
    bf16* __restrict__ outBf, float* __restrict__ outF) {
  __shared__ __align__(16) bf16 Alds[2][128][40];
  __shared__ __align__(16) bf16 Blds[2][128][40];

  const int tid = threadIdx.x;
  const int m0 = blockIdx.x * 128;
  const int n0 = blockIdx.y * 128;
  const int wave = tid >> 5, lane = tid & 31;
  const int wm = wave >> 1, wn = wave & 1;
  const int half = lane >> 4, lm = lane & 15;

  const v8f zero = {0.f, 0.f, 0.f, 0.f, 0.f, 0.f, 0.f, 0.f};
  v8f acc[2][4];
#pragma unroll
  for (int mi = 0; mi < 2; ++mi)
#pragma unroll
    for (int ni = 0; ni < 4; ++ni) acc[mi][ni] = zero;

#if TDM_MODE
  if (wave == 0) {
    tdm_load_2d(&A[(size_t)m0 * D_], &Alds[0][0][0], 32, 128, D_, 3, 3);
    tdm_load_2d(&Wt[(size_t)n0 * D_], &Blds[0][0][0], 32, 128, D_, 3, 3);
  }
#elif ASYNC_MODE
  async_stage_tile<128, 4, 40, 256>(&A[(size_t)m0 * D_], D_, &Alds[0][0][0], tid);
  async_stage_tile<128, 4, 40, 256>(&Wt[(size_t)n0 * D_], D_, &Blds[0][0][0], tid);
#endif
  int cur = 0;
  for (int k0 = 0; k0 < D_; k0 += 32) {
#if TDM_MODE
    if (wave == 0) wait_tensor0();  // current buffers landed
    __syncthreads();                // publish buf[cur]; prior reads of buf[cur^1] retired
    if (wave == 0 && (k0 + 32) < D_) {
      tdm_load_2d(&A[(size_t)m0 * D_ + k0 + 32], &Alds[cur ^ 1][0][0], 32, 128, D_, 3, 3);
      tdm_load_2d(&Wt[(size_t)n0 * D_ + k0 + 32], &Blds[cur ^ 1][0][0], 32, 128, D_, 3, 3);
    }
#elif ASYNC_MODE
    wait_async0();
    __syncthreads();
    if ((k0 + 32) < D_) {
      async_stage_tile<128, 4, 40, 256>(&A[(size_t)m0 * D_ + k0 + 32], D_,
                                        &Alds[cur ^ 1][0][0], tid);
      async_stage_tile<128, 4, 40, 256>(&Wt[(size_t)n0 * D_ + k0 + 32], D_,
                                        &Blds[cur ^ 1][0][0], tid);
    }
#else
    __syncthreads();
    for (int i = tid; i < 128 * 4; i += 256) {
      const int r = i >> 2, c = (i & 3) << 3;
      *(bf8v*)&Alds[cur][r][c] = *(const bf8v*)&A[(size_t)(m0 + r) * D_ + k0 + c];
      *(bf8v*)&Blds[cur][r][c] = *(const bf8v*)&Wt[(size_t)(n0 + r) * D_ + k0 + c];
    }
    __syncthreads();
#endif
    // Fragments: each = two contiguous 16B LDS loads (documented VGPR layouts)
    v16bf a[2], b[4];
#pragma unroll
    for (int mi = 0; mi < 2; ++mi) {
      const int rA = wm * 32 + mi * 16 + lm;
      a[mi] = ld_frag(&Alds[cur][rA][half * 8], &Alds[cur][rA][16 + half * 8]);
    }
#pragma unroll
    for (int ni = 0; ni < 4; ++ni) {
      const int nr = wn * 64 + ni * 16 + lm;
      b[ni] = ld_frag(&Blds[cur][nr][half * 8], &Blds[cur][nr][16 + half * 8]);
    }
#pragma unroll
    for (int mi = 0; mi < 2; ++mi)
#pragma unroll
      for (int ni = 0; ni < 4; ++ni)
        acc[mi][ni] = __builtin_amdgcn_wmma_f32_16x16x32_bf16(
            false, a[mi], false, b[ni], (short)0, acc[mi][ni], false, false);
    cur ^= 1;
  }

  // epilogue: C/D layout VGPR r -> row (r + 8*half), col = lane%16
#pragma unroll
  for (int mi = 0; mi < 2; ++mi) {
#pragma unroll
    for (int ni = 0; ni < 4; ++ni) {
#pragma unroll
      for (int r = 0; r < 8; ++r) {
        const int gm = m0 + wm * 32 + mi * 16 + half * 8 + r;
        const int gn = n0 + wn * 64 + ni * 16 + lm;
        float v = acc[mi][ni][r];
        if constexpr (MODE == 0) {
          v = v * sigm(v);  // SiLU
          const int b_ = gm >> 12, l = gm & (L_ - 1);
          const int h = gn >> 6, hd = gn & 63;
          outBf[(((size_t)(b_ * H_ + h)) * L_ + l) * HD_ + hd] = (bf16)v;
        } else {
          outF[(size_t)gm * D_ + gn] = v;
        }
      }
    }
  }
}

// ---------------------------------------------------------------------------
// Small projections: x[b,l,:] @ {Wb,Wfd,Wsd,Wfg,Wsg} (D x 16 each) + sigmoids.
// ---------------------------------------------------------------------------
__global__ __launch_bounds__(128) void smallproj(
    const bf16* __restrict__ x, const float* __restrict__ Wb,
    const float* __restrict__ Wfd, const float* __restrict__ fdb,
    const float* __restrict__ Wsd, const float* __restrict__ sdb,
    const float* __restrict__ Wfg, const float* __restrict__ Wsg,
    float* __restrict__ beta, float* __restrict__ fd, float* __restrict__ sd,
    float* __restrict__ fg, float* __restrict__ sg) {
  __shared__ float xs[D_];
  const int m   = blockIdx.x;  // b*L + l
  const int tid = threadIdx.x;
  for (int i = tid; i < D_; i += 128) xs[i] = (float)x[(size_t)m * D_ + i];
  __syncthreads();
  if (tid < 80) {
    const int w = tid / 16, h = tid & 15;
    const float* W = (w == 0) ? Wb : (w == 1) ? Wfd : (w == 2) ? Wsd : (w == 3) ? Wfg : Wsg;
    float acc = 0.f;
#pragma unroll 8
    for (int k = 0; k < D_; ++k) acc += xs[k] * W[k * H_ + h];
    const int b = m >> 12, l = m & (L_ - 1);
    const size_t oi = ((size_t)(b * H_ + h)) * L_ + l;
    if (w == 0)      beta[oi] = sigm(acc);
    else if (w == 1) fd[oi]   = sigm(acc + fdb[h]);
    else if (w == 2) sd[oi]   = sigm(acc + sdb[h]);
    else if (w == 3) fg[oi]   = sigm(acc);
    else             sg[oi]   = sigm(acc);
  }
}

// ---------------------------------------------------------------------------
// Per-(b,h,l) prep: beta scale, psi/surprise, slow streams, decay mods,
// combine coefficients.  One wave32 per position.
// ---------------------------------------------------------------------------
__global__ __launch_bounds__(256) void prep(
    bf16* __restrict__ kq, bf16* __restrict__ vq,
    bf16* __restrict__ kslow, bf16* __restrict__ vslow,
    const float* __restrict__ beta, const float* __restrict__ fd,
    const float* __restrict__ sd, const float* __restrict__ fg,
    const float* __restrict__ sg, float* __restrict__ fmod,
    float* __restrict__ smod, float* __restrict__ cf, float* __restrict__ cs) {
  const int gwave = (blockIdx.x * 256 + threadIdx.x) >> 5;  // bh*L + l
  const int lane  = threadIdx.x & 31;
  const size_t rb = (size_t)gwave * HD_;

  const float b  = beta[gwave];
  float k0 = (float)kq[rb + lane] * b;
  float k1 = (float)kq[rb + lane + 32] * b;
  float v0 = (float)vq[rb + lane] * b;
  float v1 = (float)vq[rb + lane + 32] * b;

  const float kk = wred(k0 * k0 + k1 * k1);
  const float vv = wred(v0 * v0 + v1 * v1);
  const float kv = wred(k0 * v0 + k1 * v1);
  const float dot = kv / ((__builtin_sqrtf(kk) + 1e-8f) * (__builtin_sqrtf(vv) + 1e-8f));
  const float psi = sigm(3.0f * __builtin_fabsf(dot));
  const float sur = sigm(10.0f * (psi - 0.5f));

  kq[rb + lane]         = (bf16)k0;
  kq[rb + lane + 32]    = (bf16)k1;
  vq[rb + lane]         = (bf16)v0;
  vq[rb + lane + 32]    = (bf16)v1;
  kslow[rb + lane]      = (bf16)(k0 * sur);
  kslow[rb + lane + 32] = (bf16)(k1 * sur);
  vslow[rb + lane]      = (bf16)(v0 * sur);
  vslow[rb + lane + 32] = (bf16)(v1 * sur);

  if (lane == 0) {
    fmod[gwave] = fd[gwave] * (1.0f - 0.1f * psi);
    smod[gwave] = sd[gwave] * (1.0f - 0.05f * psi);
    const float alpha = 0.5f + 0.3f * psi;
    cf[gwave] = alpha * fg[gwave];
    cs[gwave] = (1.0f - alpha) * sg[gwave];
  }
}

// ---------------------------------------------------------------------------
// Per-chunk decay scale: mean(decay_mod over chunk) ^ 128  (7 squarings).
// ---------------------------------------------------------------------------
__global__ __launch_bounds__(256) void chunkscale(
    const float* __restrict__ fmod, const float* __restrict__ smod,
    float* __restrict__ fscale, float* __restrict__ sscale) {
  const int gwave = (blockIdx.x * 256 + threadIdx.x) >> 5;  // bh*NCHUNK + c
  const int lane  = threadIdx.x & 31;
  const int bh = gwave >> 5, c = gwave & (NCHUNK - 1);
  const size_t base = (size_t)bh * L_ + (size_t)c * CHUNK_;
  float sf = 0.f, ss = 0.f;
#pragma unroll
  for (int i = lane; i < CHUNK_; i += 32) {
    sf += fmod[base + i];
    ss += smod[base + i];
  }
  sf = wred(sf) * (1.0f / CHUNK_);
  ss = wred(ss) * (1.0f / CHUNK_);
#pragma unroll
  for (int i = 0; i < 7; ++i) { sf *= sf; ss *= ss; }  // x^128
  if (lane == 0) {
    fscale[c * BH + bh] = sf;
    sscale[c * BH + bh] = ss;
  }
}

// ---------------------------------------------------------------------------
// Chunked recurrence.  One 128-thr block per (pass,b,h).
// q staged linearly via TDM; k,v staged TRANSPOSED (4x4 register transpose);
// state spilled transposed (sT[e][d]).
// ---------------------------------------------------------------------------
__global__ __launch_bounds__(128) void recurrence(
    const bf16* __restrict__ Q, const bf16* __restrict__ Kf,
    const bf16* __restrict__ Vf, const bf16* __restrict__ Ks,
    const bf16* __restrict__ Vs, const float* __restrict__ fscale,
    const float* __restrict__ sscale, const float* __restrict__ cf,
    const float* __restrict__ cs, float* __restrict__ ofast,
    float* __restrict__ oslow) {
  __shared__ __align__(16) bf16 qlds[CHUNK_][HD_ + 8];   // 128 x 72
  __shared__ __align__(16) bf16 kT[HD_][CHUNK_ + 8];     // 64 x 136  (kT[e][l])
  __shared__ __align__(16) bf16 vT[HD_][CHUNK_ + 8];     // 64 x 136  (vT[d][l])
  __shared__ __align__(16) bf16 sT[HD_][HD_ + 8];        // 64 x 72   (sT[e][d])

  const int pass = blockIdx.x >> 5;  // BH == 32
  const int bh   = blockIdx.x & 31;
  const bf16* Kp  = pass ? Ks : Kf;
  const bf16* Vp  = pass ? Vs : Vf;
  const float* SC = pass ? sscale : fscale;
  const float* CO = pass ? cs : cf;
  float* OUT      = pass ? oslow : ofast;
  const size_t base = (size_t)bh * L_ * HD_;

  const int tid  = threadIdx.x;
  const int wave = tid >> 5, lane = tid & 31;
  const int half = lane >> 4, lm = lane & 15;
  const int swm = wave >> 1, swn = wave & 1;

  const v8f zero = {0.f, 0.f, 0.f, 0.f, 0.f, 0.f, 0.f, 0.f};
  v8f st[2][2] = {{zero, zero}, {zero, zero}};

  for (int c = 0; c < NCHUNK; ++c) {
    const int l0 = c * CHUNK_;
#if TDM_MODE
    if (wave == 0)  // async q tile; pad 64 -> 72 elems per row
      tdm_load_2d(&Q[base + (size_t)l0 * HD_], &qlds[0][0], HD_, CHUNK_, HD_, 4, 3);
#elif ASYNC_MODE
    async_stage_tile<CHUNK_, 8, HD_ + 8, 128>(&Q[base + (size_t)l0 * HD_], HD_,
                                              &qlds[0][0], tid);
#else
    for (int i = tid; i < CHUNK_ * 8; i += 128) {
      const int r = i >> 3, col = (i & 7) << 3;
      *(bf8v*)&qlds[r][col] = *(const bf8v*)&Q[base + (size_t)(l0 + r) * HD_ + col];
    }
#endif
    // stage k,v transposed: 4x4 register transpose, vector LDS stores
    for (int sb = tid; sb < 512; sb += 128) {
      const int lb = (sb >> 4) << 2;
      const int hb = (sb & 15) << 2;
      const size_t g = base + (size_t)(l0 + lb) * HD_ + hb;
      {
        bf4 r0 = *(const bf4*)&Kp[g];
        bf4 r1 = *(const bf4*)&Kp[g + HD_];
        bf4 r2 = *(const bf4*)&Kp[g + 2 * HD_];
        bf4 r3 = *(const bf4*)&Kp[g + 3 * HD_];
        bf4 c0 = {r0[0], r1[0], r2[0], r3[0]};
        bf4 c1 = {r0[1], r1[1], r2[1], r3[1]};
        bf4 c2 = {r0[2], r1[2], r2[2], r3[2]};
        bf4 c3 = {r0[3], r1[3], r2[3], r3[3]};
        *(bf4*)&kT[hb + 0][lb] = c0;
        *(bf4*)&kT[hb + 1][lb] = c1;
        *(bf4*)&kT[hb + 2][lb] = c2;
        *(bf4*)&kT[hb + 3][lb] = c3;
      }
      {
        bf4 r0 = *(const bf4*)&Vp[g];
        bf4 r1 = *(const bf4*)&Vp[g + HD_];
        bf4 r2 = *(const bf4*)&Vp[g + 2 * HD_];
        bf4 r3 = *(const bf4*)&Vp[g + 3 * HD_];
        bf4 c0 = {r0[0], r1[0], r2[0], r3[0]};
        bf4 c1 = {r0[1], r1[1], r2[1], r3[1]};
        bf4 c2 = {r0[2], r1[2], r2[2], r3[2]};
        bf4 c3 = {r0[3], r1[3], r2[3], r3[3]};
        *(bf4*)&vT[hb + 0][lb] = c0;
        *(bf4*)&vT[hb + 1][lb] = c1;
        *(bf4*)&vT[hb + 2][lb] = c2;
        *(bf4*)&vT[hb + 3][lb] = c3;
      }
    }
#if TDM_MODE
    if (wave == 0) wait_tensor0();
#elif ASYNC_MODE
    wait_async0();
#endif
    __syncthreads();

    const float s = SC[c * BH + bh];
#pragma unroll
    for (int mi = 0; mi < 2; ++mi)
#pragma unroll
      for (int ni = 0; ni < 2; ++ni)
#pragma unroll
        for (int r = 0; r < 8; ++r) st[mi][ni][r] *= s;

    // state += v^T k   (M = d 64, N = e 64, K = l 128)
#pragma unroll
    for (int kk = 0; kk < 4; ++kk) {
      const int lb = kk * 32;
      v16bf a[2], b[2];
#pragma unroll
      for (int mi = 0; mi < 2; ++mi) {
        const int dcol = swm * 32 + mi * 16 + lm;
        a[mi] = ld_frag(&vT[dcol][lb + half * 8], &vT[dcol][lb + 16 + half * 8]);
      }
#pragma unroll
      for (int ni = 0; ni < 2; ++ni) {
        const int ecol = swn * 32 + ni * 16 + lm;
        b[ni] = ld_frag(&kT[ecol][lb + half * 8], &kT[ecol][lb + 16 + half * 8]);
      }
#pragma unroll
      for (int mi = 0; mi < 2; ++mi)
#pragma unroll
        for (int ni = 0; ni < 2; ++ni)
          st[mi][ni] = __builtin_amdgcn_wmma_f32_16x16x32_bf16(
              false, a[mi], false, b[ni], (short)0, st[mi][ni], false, false);
    }

    // spill state TRANSPOSED: sT[e][d]  (B operand for q@state is contiguous-in-d)
#pragma unroll
    for (int mi = 0; mi < 2; ++mi)
#pragma unroll
      for (int ni = 0; ni < 2; ++ni)
#pragma unroll
        for (int r = 0; r < 8; ++r)
          sT[swn * 32 + ni * 16 + lm][swm * 32 + mi * 16 + half * 8 + r] =
              (bf16)st[mi][ni][r];
    __syncthreads();

    // o = q @ state  (M = l 128, N = e 64, K = d 64); wave owns 32 rows
    v8f oc[2][4];
#pragma unroll
    for (int mi = 0; mi < 2; ++mi)
#pragma unroll
      for (int ni = 0; ni < 4; ++ni) oc[mi][ni] = zero;
#pragma unroll
    for (int kk = 0; kk < 2; ++kk) {
      const int db = kk * 32;
      v16bf a[2], b[4];
#pragma unroll
      for (int mi = 0; mi < 2; ++mi) {
        const int lr = wave * 32 + mi * 16 + lm;
        a[mi] = ld_frag(&qlds[lr][db + half * 8], &qlds[lr][db + 16 + half * 8]);
      }
#pragma unroll
      for (int ni = 0; ni < 4; ++ni) {
        const int ec = ni * 16 + lm;
        b[ni] = ld_frag(&sT[ec][db + half * 8], &sT[ec][db + 16 + half * 8]);
      }
#pragma unroll
      for (int mi = 0; mi < 2; ++mi)
#pragma unroll
        for (int ni = 0; ni < 4; ++ni)
          oc[mi][ni] = __builtin_amdgcn_wmma_f32_16x16x32_bf16(
              false, a[mi], false, b[ni], (short)0, oc[mi][ni], false, false);
    }

    // fused gate*alpha coefficient + store
#pragma unroll
    for (int mi = 0; mi < 2; ++mi) {
#pragma unroll
      for (int r = 0; r < 8; ++r) {
        const int lg = l0 + wave * 32 + mi * 16 + half * 8 + r;
        const float coef = CO[(size_t)bh * L_ + lg];
#pragma unroll
        for (int ni = 0; ni < 4; ++ni)
          OUT[base + (size_t)lg * HD_ + ni * 16 + lm] = oc[mi][ni][r] * coef;
      }
    }
    __syncthreads();
  }
}

// ---------------------------------------------------------------------------
// Combine fast+slow streams, transpose heads back: [B,H,L,HD] -> [B,L,D] bf16
// ---------------------------------------------------------------------------
__global__ void combine(const float* __restrict__ of, const float* __restrict__ os,
                        bf16* __restrict__ obf) {
  const size_t total = (size_t)M_ * D_;
  for (size_t i = (size_t)blockIdx.x * blockDim.x + threadIdx.x; i < total;
       i += (size_t)gridDim.x * blockDim.x) {
    const size_t bh = i >> 18;            // L_*HD_ = 262144
    const size_t rem = i & 262143;
    const size_t l = rem >> 6, hd = rem & 63;
    const size_t b = bh >> 4, h = bh & 15;
    obf[(b * L_ + l) * D_ + h * HD_ + hd] = (bf16)(of[i] + os[i]);
  }
}

// ---------------------------------------------------------------------------
extern "C" void kernel_launch(void* const* d_in, const int* in_sizes, int n_in,
                              void* d_out, int out_size, void* d_ws, size_t ws_size,
                              hipStream_t stream) {
  (void)in_sizes; (void)n_in; (void)out_size; (void)ws_size;
  const float* x   = (const float*)d_in[0];
  const float* Wq  = (const float*)d_in[1];
  const float* Wk  = (const float*)d_in[2];
  const float* Wv  = (const float*)d_in[3];
  const float* Wb  = (const float*)d_in[4];
  const float* Wfd = (const float*)d_in[5];
  const float* fdb = (const float*)d_in[6];
  const float* Wsd = (const float*)d_in[7];
  const float* sdb = (const float*)d_in[8];
  const float* Wfg = (const float*)d_in[9];
  const float* Wsg = (const float*)d_in[10];
  const float* Wo  = (const float*)d_in[11];

  char* ws = (char*)d_ws;
  size_t off = 0;
  auto alloc = [&](size_t bytes) -> char* {
    char* p = ws + off;
    off += (bytes + 255) & ~(size_t)255;
    return p;
  };

  bf16* x_bf  = (bf16*)alloc((size_t)M_ * D_ * 2);
  bf16* wqT   = (bf16*)alloc((size_t)D_ * D_ * 2);
  bf16* wkT   = (bf16*)alloc((size_t)D_ * D_ * 2);
  bf16* wvT   = (bf16*)alloc((size_t)D_ * D_ * 2);
  bf16* woT   = (bf16*)alloc((size_t)D_ * D_ * 2);
  bf16* q_bf  = (bf16*)alloc((size_t)M_ * D_ * 2);
  bf16* k_bf  = (bf16*)alloc((size_t)M_ * D_ * 2);
  bf16* v_bf  = (bf16*)alloc((size_t)M_ * D_ * 2);
  bf16* ks_bf = (bf16*)alloc((size_t)M_ * D_ * 2);
  bf16* vs_bf = (bf16*)alloc((size_t)M_ * D_ * 2);
  float* beta = (float*)alloc((size_t)BH * L_ * 4);
  float* fd   = (float*)alloc((size_t)BH * L_ * 4);
  float* sd   = (float*)alloc((size_t)BH * L_ * 4);
  float* fg   = (float*)alloc((size_t)BH * L_ * 4);
  float* sg   = (float*)alloc((size_t)BH * L_ * 4);
  float* fmod = (float*)alloc((size_t)BH * L_ * 4);
  float* smod = (float*)alloc((size_t)BH * L_ * 4);
  float* cf   = (float*)alloc((size_t)BH * L_ * 4);
  float* cs   = (float*)alloc((size_t)BH * L_ * 4);
  float* fscale = (float*)alloc((size_t)NCHUNK * BH * 4);
  float* sscale = (float*)alloc((size_t)NCHUNK * BH * 4);
  float* ofast  = (float*)alloc((size_t)M_ * D_ * 4);
  float* oslow  = (float*)alloc((size_t)M_ * D_ * 4);
  bf16* o_bf    = (bf16*)alloc((size_t)M_ * D_ * 2);

  cvt_f32_to_bf16<<<4096, 256, 0, stream>>>(x, x_bf, M_ * D_);
  dim3 tg(D_ / 32, D_ / 32);
  cvt_transpose_bf16<<<tg, 256, 0, stream>>>(Wq, wqT);
  cvt_transpose_bf16<<<tg, 256, 0, stream>>>(Wk, wkT);
  cvt_transpose_bf16<<<tg, 256, 0, stream>>>(Wv, wvT);
  cvt_transpose_bf16<<<tg, 256, 0, stream>>>(Wo, woT);

  dim3 gg(M_ / 128, D_ / 128);
  gemm_wmma<0><<<gg, 256, 0, stream>>>(x_bf, wqT, q_bf, nullptr);
  gemm_wmma<0><<<gg, 256, 0, stream>>>(x_bf, wkT, k_bf, nullptr);
  gemm_wmma<0><<<gg, 256, 0, stream>>>(x_bf, wvT, v_bf, nullptr);

  smallproj<<<M_, 128, 0, stream>>>(x_bf, Wb, Wfd, fdb, Wsd, sdb, Wfg, Wsg,
                                    beta, fd, sd, fg, sg);

  prep<<<(BH * L_) / 8, 256, 0, stream>>>(k_bf, v_bf, ks_bf, vs_bf, beta, fd, sd,
                                          fg, sg, fmod, smod, cf, cs);

  chunkscale<<<(BH * NCHUNK) / 8, 256, 0, stream>>>(fmod, smod, fscale, sscale);

  recurrence<<<2 * BH, 128, 0, stream>>>(q_bf, k_bf, v_bf, ks_bf, vs_bf, fscale,
                                         sscale, cf, cs, ofast, oslow);

  combine<<<4096, 256, 0, stream>>>(ofast, oslow, o_bf);

  gemm_wmma<1><<<gg, 256, 0, stream>>>(o_bf, woT, nullptr, (float*)d_out);
}
